// WaveletVariableLRSpatioTemporalGNN_40037685133383
// MI455X (gfx1250) — matrix operand order repaired and checked by reading, not compiled, not run on hardware
//
#include <hip/hip_runtime.h>
#include <hip/hip_bf16.h>

typedef __bf16 bf16;
typedef __attribute__((ext_vector_type(16))) __bf16 v16bf;
typedef __attribute__((ext_vector_type(8)))  __bf16 v8bf;
typedef __attribute__((ext_vector_type(8)))  float  v8f;

#define Bdim 4
#define Tdim 8
#define Nn   4000
#define Edim 16000
#define Hd   128
#define HCd  128
#define THd  256
#define WHd  64
#define SUBd 21

// Fragment layout maps (ISA 7.12.2, wave32, 16-bit A/B for 16x16x32):
//  A (16x32, row m = lane%16):  kk in [0,32): lane_half = (kk>>3)&1,
//      elem j_e = 8*(kk>>4) + (kk&7)   -> 8 consecutive kk = 8 consecutive elems
//  B (32x16, col n = lane%16):  kk in [0,32): lane_half = (kk>>3)&1, elem j_e = 8*(kk>>4) + (kk&7)

// ---------------------------------------------------------------------------
// fp32 [K,N] -> bf16 fragment-major [ceil(N/64)*4][ceil(K/32)][32 lanes][16],
// zero-padded. Every weight goes through this so GEMM B-fragments are direct
// coalesced global_load_b128 pairs (no LDS staging for B).
// ---------------------------------------------------------------------------
__global__ void repack_w_kernel(const float* __restrict__ src, bf16* __restrict__ dst,
                                int K, int N, int nKc, int total) {
  int i = blockIdx.x * blockDim.x + threadIdx.x;
  if (i >= total) return;
  int j_e  = i & 15;
  int lane = (i >> 4) & 31;
  int rr   = i >> 9;            // ct * nKc + kc
  int kc   = rr % nKc;
  int ct   = rr / nKc;
  int m = lane & 15, half = lane >> 4;
  int kk = ((j_e >> 3) << 4) | (half << 3) | (j_e & 7);   // B map inverse
  int k = kc * 32 + kk;
  int col = ct * 16 + m;
  dst[i] = (k < K && col < N) ? (bf16)src[(size_t)k * N + col] : (bf16)0.f;
}

// ---------------------------------------------------------------------------
// Generic WMMA GEMM:  C = act(A1 @ W1 + A2 @ W2 + bias)
// A: fp32 row-major [M,K]; W: fragment-major bf16 (repacked); C: fp32 [M,N]
// Block = 128 threads (4 waves), 64x64 tile. A staged fragment-major in LDS
// (vectorized float4 -> ds_store_b128 fast path); B read directly from global.
// ---------------------------------------------------------------------------
__global__ __launch_bounds__(128) void gemm_wmma(
    const float* __restrict__ A1, int K1, const bf16* __restrict__ W1,
    const float* __restrict__ A2, int K2, const bf16* __restrict__ W2,
    const float* __restrict__ bias, float* __restrict__ C,
    int M, int N, int act)
{
  __shared__ __attribute__((aligned(32))) bf16 As2[4][32][16];  // [wave][lane][elem]

  const int tid  = threadIdx.x;
  const int lane = tid & 31;
  const int w    = tid >> 5;
  const int row0 = blockIdx.y * 64;
  const int col0 = blockIdx.x * 64;
  const int m    = lane & 15;
  const int half = lane >> 4;

  v8f acc[4];
  #pragma unroll
  for (int nt = 0; nt < 4; ++nt)
    #pragma unroll
    for (int r = 0; r < 8; ++r) acc[nt][r] = 0.f;

  for (int chain = 0; chain < 2; ++chain) {
    const float* A = chain ? A2 : A1;
    const bf16*  W = chain ? W2 : W1;
    const int    K = chain ? K2 : K1;
    if (A == nullptr) continue;
    const int nKc = (K + 31) >> 5;
    const bool fastM = (row0 + 64 <= M) && ((K & 3) == 0);

    for (int kc = 0; kc < nKc; ++kc) {
      const int k0 = kc << 5;
      if (fastM && (k0 + 32 <= K)) {
        // fast path: 8 consecutive floats per thread -> one 16B LDS store
        #pragma unroll
        for (int it = 0; it < 2; ++it) {
          int id  = tid + it * 128;          // 256 chunks: (row, k-octet)
          int r   = id >> 2, oct = id & 3;
          const float* srcp = A + (size_t)(row0 + r) * K + k0 + oct * 8;
          float4 p0 = *(const float4*)srcp;
          float4 p1 = *(const float4*)(srcp + 4);
          v8bf vv;
          vv[0] = (bf16)p0.x; vv[1] = (bf16)p0.y; vv[2] = (bf16)p0.z; vv[3] = (bf16)p0.w;
          vv[4] = (bf16)p1.x; vv[5] = (bf16)p1.y; vv[6] = (bf16)p1.z; vv[7] = (bf16)p1.w;
          int ln = ((oct & 1) << 4) | (r & 15);
          *(v8bf*)(&As2[r >> 4][ln][(oct >> 1) * 8]) = vv;
        }
      } else {
        // guarded slow path (K remainder / ragged M)
        for (int i = tid; i < 64 * 32; i += 128) {
          int r = i >> 5, kk = i & 31;
          int gr = row0 + r, gk = k0 + kk;
          float v = (gr < M && gk < K) ? A[(size_t)gr * K + gk] : 0.f;
          int ln = (((kk >> 3) & 1) << 4) | (r & 15);
          int je = ((kk >> 4) << 3) | (kk & 7);
          As2[r >> 4][ln][je] = (bf16)v;
        }
      }
      __syncthreads();

      const v16bf a = *(const v16bf*)(&As2[w][lane][0]);
      #pragma unroll
      for (int nt = 0; nt < 4; ++nt) {
        int ct = (col0 >> 4) + nt;
        const v16bf bfr = *(const v16bf*)(W + (((size_t)ct * nKc + kc) << 9) + (lane << 4));
        acc[nt] = __builtin_amdgcn_wmma_f32_16x16x32_bf16(
            false, a, false, bfr, (short)0, acc[nt], false, false);
      }
      __syncthreads();
    }
  }

  // epilogue: C/D layout — lane = column, VGPR r -> row r + 8*half
  #pragma unroll
  for (int nt = 0; nt < 4; ++nt) {
    int gc = col0 + nt * 16 + m;
    if (gc >= N) continue;
    float bv = bias ? bias[gc] : 0.f;
    #pragma unroll
    for (int r = 0; r < 8; ++r) {
      int gr = row0 + w * 16 + 8 * half + r;
      if (gr < M) {
        float v = acc[nt][r] + bv;
        if (act) v = fmaxf(v, 0.f);
        C[(size_t)gr * N + gc] = v;
      }
    }
  }
}

// ---------------------------------------------------------------------------
// degree count (atomic), edge message scatter, mean-divide
// ---------------------------------------------------------------------------
__device__ __forceinline__ void atomAddF(float* p, float v) {
  unsafeAtomicAdd(p, v);   // lowers to global_atomic_add_f32
}

__global__ void deg_count_kernel(const int* __restrict__ dst, float* __restrict__ deg, int E) {
  int i = blockIdx.x * blockDim.x + threadIdx.x;
  if (i < E) atomAddF(&deg[dst[i]], 1.f);
}

__global__ void msg_scatter_kernel(const float* __restrict__ hs, const float* __restrict__ hd,
                                   const float* __restrict__ ef,
                                   const int* __restrict__ src, const int* __restrict__ dst,
                                   float* __restrict__ agg, int BT) {
  size_t t = (size_t)blockIdx.x * blockDim.x + threadIdx.x;
  size_t total = (size_t)BT * Edim * Hd;
  if (t >= total) return;
  int j = (int)(t & (Hd - 1));
  size_t te = t >> 7;
  int e  = (int)(te % Edim);
  int bt = (int)(te / Edim);
  int s = src[e], d = dst[e];
  float v = hs[((size_t)bt * Nn + s) * Hd + j] +
            hd[((size_t)bt * Nn + d) * Hd + j] +
            ef[(size_t)e * Hd + j];
  v = fmaxf(v, 0.f);
  atomAddF(&agg[((size_t)bt * Nn + d) * Hd + j], v);
}

__global__ void agg_div_kernel(float* __restrict__ agg, const float* __restrict__ deg, int BT) {
  size_t t = (size_t)blockIdx.x * blockDim.x + threadIdx.x;
  size_t total = (size_t)BT * Nn * Hd;
  if (t >= total) return;
  size_t rn = t >> 7;
  int n = (int)(rn % Nn);
  agg[t] = agg[t] / fmaxf(deg[n], 1.f);
}

// ---------------------------------------------------------------------------
// residual add + LayerNorm, one row per block (blockDim.x == D, D = 64/128)
// ---------------------------------------------------------------------------
__global__ void add_ln_kernel(const float* __restrict__ X, const float* __restrict__ R,
                              float* __restrict__ O,
                              const float* __restrict__ g, const float* __restrict__ b, int D) {
  __shared__ float red[256];
  int row = blockIdx.x;
  int j = threadIdx.x;
  float s = X[(size_t)row * D + j] + (R ? R[(size_t)row * D + j] : 0.f);
  red[j] = s; __syncthreads();
  for (int st = D >> 1; st > 0; st >>= 1) { if (j < st) red[j] += red[j + st]; __syncthreads(); }
  float mean = red[0] / (float)D; __syncthreads();
  float dm = s - mean;
  red[j] = dm * dm; __syncthreads();
  for (int st = D >> 1; st > 0; st >>= 1) { if (j < st) red[j] += red[j + st]; __syncthreads(); }
  float var = red[0] / (float)D;
  O[(size_t)row * D + j] = dm * rsqrtf(var + 1e-5f) * g[j] + b[j];
}

// ---------------------------------------------------------------------------
// LSTM: 16 nodes per block (256 thr / 8 waves). Per timestep, WMMA computes
// z = seq@Wih + h@Whh (two K=128 chains into 16x512). seq/h tiles live in LDS
// fragment-major; weights pre-repacked fragment-major in global. z is spilled
// transposed so each accumulator is one contiguous v8f (2x ds_store_b128).
// ---------------------------------------------------------------------------
__global__ __launch_bounds__(256) void lstm_kernel(
    const float* __restrict__ h,        // [B*T*Nn, 128], row = bt*Nn + n
    const bf16* __restrict__ WihPk,     // fragment-major [32][4][32][16]
    const bf16* __restrict__ WhhPk,     // fragment-major [32][4][32][16]
    const float* __restrict__ bias,     // [512]
    float* __restrict__ feat)           // [B*Nn, 128]
{
  __shared__ __attribute__((aligned(32))) bf16 sSeqF[4][32][16];  // [kc][lane][elem]
  __shared__ __attribute__((aligned(32))) bf16 sHHF [4][32][16];
  __shared__ __attribute__((aligned(32))) float sZt[512][16];     // [col][node]
  __shared__ float sC [16][128];
  __shared__ float sHf[16][128];

  const int tid = threadIdx.x;
  const int lane = tid & 31;
  const int w = tid >> 5;
  const int m = lane & 15;
  const int half = lane >> 4;
  const int bn0 = blockIdx.x * 16;

  for (int i = tid; i < 4 * 32 * 16; i += 256)
    sHHF[i >> 9][(i >> 4) & 31][i & 15] = (bf16)0.f;
  for (int i = tid; i < 16 * 128; i += 256)
    sC[i >> 7][i & 127] = 0.f;
  __syncthreads();

  for (int t = 0; t < Tdim; ++t) {
    // stage seq tile: one (row, k-octet) chunk per thread, vectorized
    {
      int r = tid >> 4, oct = tid & 15;          // 16 rows x 16 octets
      int bn = bn0 + r;
      int b = bn / Nn, n = bn % Nn;
      const float* srcp = h + ((size_t)(b * Tdim + t) * Nn + n) * 128 + oct * 8;
      float4 p0 = *(const float4*)srcp;
      float4 p1 = *(const float4*)(srcp + 4);
      v8bf vv;
      vv[0] = (bf16)p0.x; vv[1] = (bf16)p0.y; vv[2] = (bf16)p0.z; vv[3] = (bf16)p0.w;
      vv[4] = (bf16)p1.x; vv[5] = (bf16)p1.y; vv[6] = (bf16)p1.z; vv[7] = (bf16)p1.w;
      int kc = oct >> 2, o2 = oct & 3;
      int ln = ((o2 & 1) << 4) | r;
      *(v8bf*)(&sSeqF[kc][ln][(o2 >> 1) * 8]) = vv;
    }
    __syncthreads();

    v8f acc[4];
    #pragma unroll
    for (int nt = 0; nt < 4; ++nt)
      #pragma unroll
      for (int r = 0; r < 8; ++r) acc[nt][r] = 0.f;

    #pragma unroll
    for (int chain = 0; chain < 2; ++chain) {
      const bf16* Wg = chain ? WhhPk : WihPk;
      #pragma unroll
      for (int kc = 0; kc < 4; ++kc) {
        const v16bf a = *(const v16bf*)(chain ? &sHHF[kc][lane][0] : &sSeqF[kc][lane][0]);
        #pragma unroll
        for (int nt = 0; nt < 4; ++nt) {
          int ct = w * 4 + nt;                       // column tile 0..31
          const v16bf bfr = *(const v16bf*)(Wg + ((((size_t)ct * 4 + kc) * 32 + lane) << 4));
          acc[nt] = __builtin_amdgcn_wmma_f32_16x16x32_bf16(
              false, a, false, bfr, (short)0, acc[nt], false, false);
        }
      }
    }

    // transposed spill: one v8f (rows 8*half..8*half+7 of this column) each
    #pragma unroll
    for (int nt = 0; nt < 4; ++nt)
      *(v8f*)(&sZt[w * 64 + nt * 16 + m][8 * half]) = acc[nt];
    __syncthreads();

    // gates (fp32), write h back in fragment-major form for next step
    for (int i = tid; i < 16 * 128; i += 256) {
      int r = i >> 7, j = i & 127;
      float zi = sZt[j][r]       + bias[j];
      float zf = sZt[128 + j][r] + bias[128 + j];
      float zg = sZt[256 + j][r] + bias[256 + j];
      float zo = sZt[384 + j][r] + bias[384 + j];
      float ig = 1.f / (1.f + __expf(-zi));
      float fg = 1.f / (1.f + __expf(-zf));
      float og = 1.f / (1.f + __expf(-zo));
      float c = fg * sC[r][j] + ig * tanhf(zg);
      float hn = og * tanhf(c);
      sC[r][j] = c;
      sHf[r][j] = hn;
      int kc = j >> 5, kk = j & 31;
      int ln = (((kk >> 3) & 1) << 4) | r;
      int je = ((kk >> 4) << 3) | (kk & 7);
      sHHF[kc][ln][je] = (bf16)hn;
    }
    __syncthreads();
  }

  for (int i = tid; i < 16 * 128; i += 256) {
    int r = i >> 7, j = i & 127;
    feat[(size_t)(bn0 + r) * 128 + j] = sHf[r][j];
  }
}

// ---------------------------------------------------------------------------
// wavelet band pooling: comps[bn, l*21+j] = fw[l] * mean_t f_l[(b,t,n), j]
// ---------------------------------------------------------------------------
__global__ void pool_kernel(const float* __restrict__ fb, const float* __restrict__ fw,
                            float* __restrict__ comps) {
  int idx = blockIdx.x * blockDim.x + threadIdx.x;
  int total = Bdim * Nn * (3 * SUBd);
  if (idx >= total) return;
  int c  = idx % (3 * SUBd);
  int bn = idx / (3 * SUBd);
  int l = c / SUBd, j = c % SUBd;
  int b = bn / Nn, n = bn % Nn;
  int t0 = (l == 0) ? (Tdim / 2) : 0;
  size_t bandOff = (size_t)l * Bdim * Tdim * Nn * SUBd;
  float s = 0.f;
  for (int t = t0; t < Tdim; ++t)
    s += fb[bandOff + ((size_t)(b * Tdim + t) * Nn + n) * SUBd + j];
  comps[(size_t)bn * (3 * SUBd) + c] = s / (float)(Tdim - t0) * fw[l];
}

// ---------------------------------------------------------------------------
// output heads
// ---------------------------------------------------------------------------
__global__ void heads_kernel(const float* __restrict__ sh,
                             const float* __restrict__ swhW, const float* __restrict__ swhb,
                             const float* __restrict__ mwdW, const float* __restrict__ mwdb,
                             const float* __restrict__ mwpW, const float* __restrict__ mwpb,
                             float* __restrict__ out, int M) {
  int r = blockIdx.x * blockDim.x + threadIdx.x;
  if (r >= M) return;
  float s0 = 0.f, s1 = 0.f, s2 = 0.f, s3 = 0.f;
  for (int j = 0; j < WHd; ++j) {
    float v = sh[(size_t)r * WHd + j];
    s0 += v * swhW[j];
    s1 += v * mwdW[j * 2 + 0];
    s2 += v * mwdW[j * 2 + 1];
    s3 += v * mwpW[j];
  }
  out[(size_t)r * 4 + 0] = s0 + swhb[0];
  out[(size_t)r * 4 + 1] = s1 + mwdb[0];
  out[(size_t)r * 4 + 2] = s2 + mwdb[1];
  out[(size_t)r * 4 + 3] = s3 + mwpb[0];
}

// ---------------------------------------------------------------------------
// host orchestration
// ---------------------------------------------------------------------------
extern "C" void kernel_launch(void* const* d_in, const int* in_sizes, int n_in,
                              void* d_out, int out_size, void* d_ws, size_t ws_size,
                              hipStream_t stream) {
  (void)in_sizes; (void)n_in; (void)out_size; (void)ws_size;

  const int BT  = Bdim * Tdim;           // 32
  const int BTN = BT * Nn;               // 128000
  const int BN  = Bdim * Nn;             // 16000

  const float* x     = (const float*)d_in[0];
  const int*   eidx  = (const int*)d_in[1];
  const int*   esrc  = eidx;
  const int*   edst  = eidx + Edim;
  const float* eattr = (const float*)d_in[2];
  auto P = [&](int i) { return (const float*)d_in[i]; };

  // bump allocator over d_ws
  char* base = (char*)d_ws;
  size_t cur = 0;
  auto allocf = [&](size_t n) -> float* {
    float* p = (float*)(base + cur); cur += ((n * 4 + 255) / 256) * 256; return p;
  };
  auto allocb = [&](size_t n) -> bf16* {
    bf16* p = (bf16*)(base + cur); cur += ((n * 2 + 255) / 256) * 256; return p;
  };

  float* f_h    = allocf((size_t)BTN * Hd);     // node features
  float* f_A    = allocf((size_t)BTN * Hd);     // hs / upd
  float* f_B    = allocf((size_t)BTN * Hd);     // hd / band hidden
  float* f_C    = allocf((size_t)BTN * Hd);     // agg / band outputs (3*BTN*21 fits)
  float* f_T64  = allocf((size_t)BTN * 64);     // enc hidden / xp
  float* f_e    = allocf((size_t)Edim * Hd);    // edge messages
  float* f_deg  = allocf(Nn);
  float* f_lstm = allocf((size_t)BN * HCd);
  float* f_cmp  = allocf((size_t)BN * 3 * SUBd);
  float* f_s1   = allocf((size_t)BN * WHd);     // proj
  float* f_s2   = allocf((size_t)BN * WHd);     // scratch
  float* f_s3   = allocf((size_t)BN * WHd);     // att
  float* f_s4   = allocf((size_t)BN * WHd);     // fused (LN)
  float* f_s5   = allocf((size_t)BN * WHd);     // wavelet out
  float* f_tf1  = allocf((size_t)BN * THd);
  float* f_tf2  = allocf((size_t)BN * THd);
  float* f_sh   = allocf((size_t)BN * WHd);

  // fragment-major weight repack (fp32 -> bf16, zero-padded)
  auto repack = [&](const float* src, int K, int N) -> bf16* {
    int nKc = (K + 31) / 32;
    int nCt = ((N + 63) / 64) * 4;
    size_t total = (size_t)nCt * nKc * 32 * 16;
    bf16* p = allocb(total);
    repack_w_kernel<<<(int)((total + 255) / 256), 256, 0, stream>>>(src, p, K, N, nKc, (int)total);
    return p;
  };

  bf16* encW1 = repack(P(3), 11, 64);
  bf16* encW2 = repack(P(5), 64, 128);
  bf16 *Wsrc[3], *Wdst[3], *We[3], *WuA[3], *WuB[3];
  for (int l = 0; l < 3; ++l) {
    int b7 = 7 + 8 * l;
    Wsrc[l] = repack(P(b7 + 0), 128, 128);
    Wdst[l] = repack(P(b7 + 1), 128, 128);
    We[l]   = repack(P(b7 + 2), 3, 128);
    WuA[l]  = repack(P(b7 + 4), 128, 128);               // W_u rows 0..127
    WuB[l]  = repack(P(b7 + 4) + 128 * 128, 128, 128);   // W_u rows 128..255
  }
  bf16* WihPk = repack(P(31), 128, 512);
  bf16* WhhPk = repack(P(32), 128, 512);
  bf16* Wp  = repack(P(34), 128, 64);
  bf16 *bW1[3], *bW2[3];
  for (int l = 0; l < 3; ++l) {
    int b37 = 37 + 4 * l;
    bW1[l] = repack(P(b37 + 0), 64, 32);
    bW2[l] = repack(P(b37 + 2), 32, SUBd);
  }
  bf16* Wfu  = repack(P(49), 63, 64);
  bf16* Wv   = repack(P(51), 64, 64);
  bf16* Wo   = repack(P(53), 64, 64);
  bf16* Wo1  = repack(P(57), 64, 64);
  bf16* Wo2  = repack(P(59), 64, 64);
  bf16* fW1a = repack(P(61), 128, 256);                  // fus_W1 rows 0..127
  bf16* fW1b = repack(P(61) + 128 * 256, 64, 256);       // fus_W1 rows 128..191
  bf16* fW2  = repack(P(63), 256, 256);
  bf16* shW  = repack(P(65), 256, 64);

  auto gemm = [&](const float* A1, int K1, const bf16* W1,
                  const float* A2, int K2, const bf16* W2,
                  const float* bias, float* Cout, int M, int N, int act) {
    dim3 g((N + 63) / 64, (M + 63) / 64), blk(128);
    gemm_wmma<<<g, blk, 0, stream>>>(A1, K1, W1, A2, K2, W2, bias, Cout, M, N, act);
  };

  // ---- encoder: h = relu(relu(x@W1+b1)@W2+b2)
  gemm(x, 11, encW1, nullptr, 0, nullptr, P(4), f_T64, BTN, 64, 1);
  gemm(f_T64, 64, encW2, nullptr, 0, nullptr, P(6), f_h, BTN, 128, 1);

  // ---- degree
  hipMemsetAsync(f_deg, 0, Nn * sizeof(float), stream);
  deg_count_kernel<<<(Edim + 255) / 256, 256, 0, stream>>>(edst, f_deg, Edim);

  // ---- 3 message-passing layers
  for (int l = 0; l < 3; ++l) {
    int b7 = 7 + 8 * l;
    gemm(f_h, 128, Wsrc[l], nullptr, 0, nullptr, nullptr, f_A, BTN, 128, 0);   // hs
    gemm(f_h, 128, Wdst[l], nullptr, 0, nullptr, nullptr, f_B, BTN, 128, 0);   // hd
    gemm(eattr, 3, We[l], nullptr, 0, nullptr, P(b7 + 3), f_e, Edim, 128, 0);  // e + b_msg
    hipMemsetAsync(f_C, 0, (size_t)BTN * Hd * sizeof(float), stream);
    {
      size_t total = (size_t)BT * Edim * Hd;
      msg_scatter_kernel<<<(int)((total + 255) / 256), 256, 0, stream>>>(
          f_A, f_B, f_e, esrc, edst, f_C, BT);
    }
    {
      size_t total = (size_t)BTN * Hd;
      agg_div_kernel<<<(int)((total + 255) / 256), 256, 0, stream>>>(f_C, f_deg, BT);
    }
    // upd = relu([h, agg] @ W_u + b_u)
    gemm(f_h, 128, WuA[l], f_C, 128, WuB[l], P(b7 + 5), f_A, BTN, 128, 1);
    // h = LN(h + upd)
    add_ln_kernel<<<BTN, 128, 0, stream>>>(f_h, f_A, f_h, P(b7 + 6), P(b7 + 7), 128);
  }

  // ---- LSTM over T per node (16 nodes per block, WMMA gate GEMMs)
  lstm_kernel<<<BN / 16, 256, 0, stream>>>(f_h, WihPk, WhhPk, P(33), f_lstm);

  // ---- wavelet processor
  gemm(f_h, 128, Wp, nullptr, 0, nullptr, P(35), f_T64, BTN, WHd, 0);   // xp
  for (int l = 0; l < 3; ++l) {
    int b37 = 37 + 4 * l;
    gemm(f_T64, 64, bW1[l], nullptr, 0, nullptr, P(b37 + 1), f_B, BTN, 32, 1);
    gemm(f_B, 32, bW2[l], nullptr, 0, nullptr, P(b37 + 3),
         f_C + (size_t)l * BTN * SUBd, BTN, SUBd, 0);
  }
  {
    int total = Bdim * Nn * 3 * SUBd;
    pool_kernel<<<(total + 255) / 256, 256, 0, stream>>>(f_C, P(36), f_cmp);
  }
  gemm(f_cmp, 63, Wfu, nullptr, 0, nullptr, P(50), f_s1, BN, 64, 0);    // proj
  gemm(f_s1, 64, Wv, nullptr, 0, nullptr, P(52), f_s2, BN, 64, 0);
  gemm(f_s2, 64, Wo, nullptr, 0, nullptr, P(54), f_s3, BN, 64, 0);      // att
  add_ln_kernel<<<BN, 64, 0, stream>>>(f_s3, nullptr, f_s4, P(55), P(56), 64);
  gemm(f_s4, 64, Wo1, nullptr, 0, nullptr, P(58), f_s2, BN, 64, 1);
  gemm(f_s2, 64, Wo2, nullptr, 0, nullptr, P(60), f_s5, BN, 64, 0);     // wv

  // ---- fusion + heads
  gemm(f_lstm, 128, fW1a, f_s5, 64, fW1b, P(62), f_tf1, BN, 256, 1);
  gemm(f_tf1, 256, fW2, nullptr, 0, nullptr, P(64), f_tf2, BN, 256, 0);
  gemm(f_tf2, 256, shW, nullptr, 0, nullptr, P(66), f_sh, BN, 64, 1);
  heads_kernel<<<(BN + 255) / 256, 256, 0, stream>>>(
      f_sh, P(67), P(68), P(69), P(70), P(71), P(72), (float*)d_out, BN);
}
